// LINKX_9285719294274
// MI455X (gfx1250) — matrix operand order repaired and checked by reading, not compiled, not run on hardware
//
#include <hip/hip_runtime.h>

typedef _Float16 v16h __attribute__((ext_vector_type(16)));
typedef _Float16 v8h  __attribute__((ext_vector_type(8)));
typedef float    v8f  __attribute__((ext_vector_type(8)));

#define H_DIM 256
#define F_DIM 128
#define C_DIM 40
#define C_PAD 48
#define BN_EPS 1e-5f

// ---------------- helpers ----------------

// dst[i] = bias[i & mask]  (init accumulator with broadcast bias; width pow2)
__global__ void k_init_bias(float* __restrict__ dst, const float* __restrict__ bias,
                            int total, int mask) {
  int i = blockIdx.x * blockDim.x + threadIdx.x;
  if (i < total) dst[i] = bias[i & mask];
}

// flat f32 -> f16 conversion
__global__ void k_f32_to_f16(const float* __restrict__ s, _Float16* __restrict__ d, int n) {
  int i = blockIdx.x * blockDim.x + threadIdx.x;
  if (i < n) d[i] = (_Float16)s[i];
}

// W [K, Nout] f32 row-major -> WT [Npad, K] f16 (zero pad rows n >= Nout)
__global__ void k_wt_transpose(const float* __restrict__ W, _Float16* __restrict__ WT,
                               int K, int Nout, int Npad) {
  int i = blockIdx.x * blockDim.x + threadIdx.x;
  int total = Npad * K;
  if (i >= total) return;
  int k = i % K;
  int n = i / K;
  WT[n * K + k] = (n < Nout) ? (_Float16)W[k * Nout + n] : (_Float16)0.0f;
}

__global__ void k_zero_f32(float* __restrict__ p, int n) {
  int i = blockIdx.x * blockDim.x + threadIdx.x;
  if (i < n) p[i] = 0.0f;
}

// edge scatter: one wave (32 lanes) per edge; gather W_edge[src] row, scale,
// scatter-add into h_a[dst] via f32 global atomics (h_a is L2-resident, 51MB)
__global__ void k_edge_scatter(const int* __restrict__ ei, const float* __restrict__ ew,
                               const float* __restrict__ We, float* __restrict__ ha, int E) {
  long tid = (long)blockIdx.x * blockDim.x + threadIdx.x;
  long e = tid >> 5;
  int lane = (int)(tid & 31);
  if (e >= E) return;
  int src = ei[e];
  int dst = ei[(long)E + e];
  float w = ew[e];
  const float* wr = We + (long)src * H_DIM;
  float* orow = ha + (long)dst * H_DIM;
#pragma unroll
  for (int c = lane; c < H_DIM; c += 32)
    atomicAdd(&orow[c], w * wr[c]);
}

// BN finalize: scale/shift from per-channel sums over M rows
__global__ void k_bn_finalize(const float* __restrict__ sums, const float* __restrict__ sumsq,
                              const float* __restrict__ gamma, const float* __restrict__ beta,
                              float* __restrict__ scale, float* __restrict__ shift, float invM) {
  int t = threadIdx.x;
  if (t < H_DIM) {
    float mu  = sums[t] * invM;
    float var = sumsq[t] * invM - mu * mu;
    float sc  = gamma[t] * rsqrtf(var + BN_EPS);
    scale[t] = sc;
    shift[t] = beta[t] - mu * sc;
  }
}

// hn16[i] = f16(h[i]*scale[c] + shift[c]),  c = i & 255
__global__ void k_norm_to_f16(const float* __restrict__ h, const float* __restrict__ scale,
                              const float* __restrict__ shift, _Float16* __restrict__ d, int n) {
  int i = blockIdx.x * blockDim.x + threadIdx.x;
  if (i < n) {
    int c = i & (H_DIM - 1);
    d[i] = (_Float16)(fmaf(h[i], scale[c], shift[c]));
  }
}

// ---------------- WMMA GEMM: register-blocked + software-pipelined ----------------
// C[M,N] = A[M,K](f16) @ B[K,N], BT16 = B^T padded [Npad,K] f16.
// One wave computes a 16 x (16*NBLK) strip with NBLK accumulators.
// K-loop is double-buffered: next k-step's A/B fragments are loaded before the
// current WMMAs issue, so loads stay in flight behind v_wmma (partial waits).
// MODE 1: outF = acc + bias + add0                     (out = h_a + h_a@Wc1 + b)
// MODE 2: outF = acc + bias ; outH = f16(outF)         (h_x)
// MODE 3: outH = f16(relu(acc + bias + add0 + add1))   (out, pre-final)
// MODE 4: outF = relu(acc + bias); atomic sums/sumsq   (h, + BN stats)
// MODE 5: outF = acc + bias, only n < Nout             (logits, ld=40)
template <int MODE, int NBLK>
__global__ void k_gemm_wmma(const _Float16* __restrict__ A16,
                            const _Float16* __restrict__ BT16,
                            const float* __restrict__ bias,
                            const float* __restrict__ add0,
                            const float* __restrict__ add1,
                            float* __restrict__ outF,
                            _Float16* __restrict__ outH,
                            float* __restrict__ sums,
                            float* __restrict__ sumsq,
                            int M, int K, int Nout, int ldOut) {
  const int lane    = threadIdx.x & 31;
  const int wave    = threadIdx.x >> 5;
  const int mBase   = (blockIdx.y * 8 + wave) * 16;
  if (mBase >= M) return;
  const int nBase0  = blockIdx.x * (16 * NBLK);
  const int nLoc    = lane & 15;
  const int halfSel = lane >> 4;                       // 0: lanes 0-15, 1: lanes 16-31

  // A fragment source: row (mBase + nLoc); per K-step, K offset k0 + hs*16
  const _Float16* aRow = A16 + (long)(mBase + nLoc) * K + halfSel * 16;
  // B fragment sources: BT row = output column, K offset k0 + hs*16
  const _Float16* bCol[NBLK];
#pragma unroll
  for (int j = 0; j < NBLK; ++j)
    bCol[j] = BT16 + (long)(nBase0 + j * 16 + nLoc) * K + halfSel * 16;

  auto loadA = [&](int k0) -> v16h {
    v8h lo = *(const v8h*)(aRow + k0);                 // K = k0 + hs*16 + [0..7]
    v8h hi = *(const v8h*)(aRow + k0 + 8);             // K = k0 + hs*16 + [8..15]
    v16h a;
#pragma unroll
    for (int i = 0; i < 8; ++i) { a[i] = lo[i]; a[i + 8] = hi[i]; }
    return a;
  };
  auto loadB = [&](int j, int k0) -> v16h {
    v8h lo = *(const v8h*)(bCol[j] + k0);              // K = k0 + hs*16 + [0..7]
    v8h hi = *(const v8h*)(bCol[j] + k0 + 8);          // K = k0 + hs*16 + [8..15]
    v16h b;
#pragma unroll
    for (int i = 0; i < 8; ++i) { b[i] = lo[i]; b[i + 8] = hi[i]; }
    return b;
  };

  v8f acc[NBLK];
#pragma unroll
  for (int j = 0; j < NBLK; ++j) acc[j] = (v8f){};

  // prologue: fragments for k=0
  v16h aCur = loadA(0);
  v16h bCur[NBLK];
#pragma unroll
  for (int j = 0; j < NBLK; ++j) bCur[j] = loadB(j, 0);

  // steady state: issue k0 loads, then WMMAs on (k0-32) fragments
  for (int k0 = 32; k0 < K; k0 += 32) {
    v16h aNxt = loadA(k0);
    v16h bNxt[NBLK];
#pragma unroll
    for (int j = 0; j < NBLK; ++j) bNxt[j] = loadB(j, k0);
#pragma unroll
    for (int j = 0; j < NBLK; ++j)
      acc[j] = __builtin_amdgcn_wmma_f32_16x16x32_f16(false, aCur, false, bCur[j],
                                                      (short)0, acc[j], false, false);
    aCur = aNxt;
#pragma unroll
    for (int j = 0; j < NBLK; ++j) bCur[j] = bNxt[j];
  }
  // epilogue k-step
#pragma unroll
  for (int j = 0; j < NBLK; ++j)
    acc[j] = __builtin_amdgcn_wmma_f32_16x16x32_f16(false, aCur, false, bCur[j],
                                                    (short)0, acc[j], false, false);

#pragma unroll
  for (int j = 0; j < NBLK; ++j) {
    const int n = nBase0 + j * 16 + nLoc;
    const float bn = (MODE == 5 && n >= Nout) ? 0.0f : bias[n];
    float ssum = 0.0f, ssq = 0.0f;
#pragma unroll
    for (int r = 0; r < 8; ++r) {
      const int m = mBase + r + halfSel * 8;           // C/D layout: M = r + (lane>=16)*8
      float v = acc[j][r] + bn;
      if (MODE == 1) v += add0[(long)m * ldOut + n];
      if (MODE == 3) v += add0[(long)m * H_DIM + n] + add1[(long)m * H_DIM + n];
      if (MODE == 3 || MODE == 4) v = fmaxf(v, 0.0f);
      if (MODE == 1 || MODE == 4) outF[(long)m * ldOut + n] = v;
      if (MODE == 2) { outF[(long)m * ldOut + n] = v; outH[(long)m * ldOut + n] = (_Float16)v; }
      if (MODE == 3) outH[(long)m * H_DIM + n] = (_Float16)v;
      if (MODE == 5) { if (n < Nout) outF[(long)m * ldOut + n] = v; }
      if (MODE == 4) { ssum += v; ssq += v * v; }
    }
    if (MODE == 4) {                                   // BN stats: 512 L2-resident counters
      atomicAdd(&sums[n], ssum);
      atomicAdd(&sumsq[n], ssq);
    }
  }
}

// ---------------- host launch ----------------

extern "C" void kernel_launch(void* const* d_in, const int* in_sizes, int n_in,
                              void* d_out, int out_size, void* d_ws, size_t ws_size,
                              hipStream_t stream) {
  const float* x       = (const float*)d_in[0];
  const int*   ei      = (const int*)  d_in[1];
  const float* ew      = (const float*)d_in[2];
  const float* W_edge  = (const float*)d_in[3];
  const float* b_edge  = (const float*)d_in[4];
  const float* W_node  = (const float*)d_in[5];
  const float* b_node  = (const float*)d_in[6];
  const float* W_c1    = (const float*)d_in[7];
  const float* b_c1    = (const float*)d_in[8];
  const float* W_c2    = (const float*)d_in[9];
  const float* b_c2    = (const float*)d_in[10];
  const float* W_f1    = (const float*)d_in[11];
  const float* b_f1    = (const float*)d_in[12];
  const float* gamma   = (const float*)d_in[13];
  const float* beta    = (const float*)d_in[14];
  const float* W_f2    = (const float*)d_in[15];
  const float* b_f2    = (const float*)d_in[16];
  float* out           = (float*)d_out;

  const int E = in_sizes[2];
  const int N = in_sizes[0] / F_DIM;
  const int NH = N * H_DIM;
  const int NF = N * F_DIM;

  // workspace carve-up (256B aligned)
  size_t off = 0;
  auto carve = [&](size_t bytes) -> void* {
    void* p = (char*)d_ws + off;
    off += (bytes + 255) & ~(size_t)255;
    return p;
  };
  float*    ha     = (float*)   carve((size_t)NH * 4);  // reused as h (post f1)
  _Float16* ha16   = (_Float16*)carve((size_t)NH * 2);  // reused as normalized h16
  _Float16* x16    = (_Float16*)carve((size_t)NF * 2);
  float*    OUT    = (float*)   carve((size_t)NH * 4);
  float*    hx     = (float*)   carve((size_t)NH * 4);
  _Float16* hx16   = (_Float16*)carve((size_t)NH * 2);
  _Float16* out16  = (_Float16*)carve((size_t)NH * 2);
  _Float16* Wc1T   = (_Float16*)carve((size_t)H_DIM * H_DIM * 2);
  _Float16* WnodeT = (_Float16*)carve((size_t)H_DIM * F_DIM * 2);
  _Float16* Wc2T   = (_Float16*)carve((size_t)H_DIM * H_DIM * 2);
  _Float16* Wf1T   = (_Float16*)carve((size_t)H_DIM * H_DIM * 2);
  _Float16* Wf2T   = (_Float16*)carve((size_t)C_PAD * H_DIM * 2);
  float*    stats  = (float*)   carve(2 * H_DIM * 4);   // sums | sumsq, contiguous
  float*    sums   = stats;
  float*    sumsq  = stats + H_DIM;
  float*    scale  = (float*)   carve(H_DIM * 4);
  float*    shift  = (float*)   carve(H_DIM * 4);

  const int TB = 256;
  const int mTiles = N / 16;                       // 3125
  const dim3 blk(TB);
  const dim3 gH(H_DIM / (16 * 4), (mTiles + 7) / 8);   // NBLK=4: grid.x = 4
  const dim3 gC(1, (mTiles + 7) / 8);                  // NBLK=3: 48-wide final GEMM

  // 1) init h_a with b_edge broadcast
  k_init_bias<<<(NH + TB - 1) / TB, blk, 0, stream>>>(ha, b_edge, NH, H_DIM - 1);

  // 2) weight transposes + x cast (tiny)
  k_wt_transpose<<<(H_DIM * H_DIM + TB - 1) / TB, blk, 0, stream>>>(W_c1,  Wc1T,   H_DIM, H_DIM, H_DIM);
  k_wt_transpose<<<(H_DIM * F_DIM + TB - 1) / TB, blk, 0, stream>>>(W_node, WnodeT, F_DIM, H_DIM, H_DIM);
  k_wt_transpose<<<(H_DIM * H_DIM + TB - 1) / TB, blk, 0, stream>>>(W_c2,  Wc2T,   H_DIM, H_DIM, H_DIM);
  k_wt_transpose<<<(H_DIM * H_DIM + TB - 1) / TB, blk, 0, stream>>>(W_f1,  Wf1T,   H_DIM, H_DIM, H_DIM);
  k_wt_transpose<<<(C_PAD * H_DIM + TB - 1) / TB, blk, 0, stream>>>(W_f2,  Wf2T,   H_DIM, C_DIM, C_PAD);
  k_f32_to_f16<<<(NF + TB - 1) / TB, blk, 0, stream>>>(x, x16, NF);

  // 3) edge gather/scatter (f32 global atomics into L2-resident h_a)
  {
    long threads = (long)E * 32;
    k_edge_scatter<<<(unsigned)((threads + TB - 1) / TB), blk, 0, stream>>>(ei, ew, W_edge, ha, E);
  }

  // 4) h_a -> f16 for GEMM input
  k_f32_to_f16<<<(NH + TB - 1) / TB, blk, 0, stream>>>(ha, ha16, NH);

  // 5) OUT = h_a + h_a@Wc1 + b_c1
  k_gemm_wmma<1, 4><<<gH, blk, 0, stream>>>(ha16, Wc1T, b_c1, ha, nullptr,
                                            OUT, nullptr, nullptr, nullptr,
                                            N, H_DIM, H_DIM, H_DIM);
  // 6) hx = x@Wnode + b_node (f32 + f16 copies)
  k_gemm_wmma<2, 4><<<gH, blk, 0, stream>>>(x16, WnodeT, b_node, nullptr, nullptr,
                                            hx, hx16, nullptr, nullptr,
                                            N, F_DIM, H_DIM, H_DIM);
  // 7) out16 = f16(relu(OUT + hx + hx@Wc2 + b_c2))
  k_gemm_wmma<3, 4><<<gH, blk, 0, stream>>>(hx16, Wc2T, b_c2, OUT, hx,
                                            nullptr, out16, nullptr, nullptr,
                                            N, H_DIM, H_DIM, H_DIM);
  // 8) BN stat counters start at zero every call (graph-replay safe)
  k_zero_f32<<<1, 2 * H_DIM, 0, stream>>>(stats, 2 * H_DIM);
  // 9) h = relu(out@Wf1 + b_f1), accumulate BN sums/sumsq  (h aliases ha)
  k_gemm_wmma<4, 4><<<gH, blk, 0, stream>>>(out16, Wf1T, b_f1, nullptr, nullptr,
                                            ha, nullptr, sums, sumsq,
                                            N, H_DIM, H_DIM, H_DIM);
  // 10) BN scale/shift
  k_bn_finalize<<<1, H_DIM, 0, stream>>>(sums, sumsq, gamma, beta, scale, shift, 1.0f / (float)N);
  // 11) normalized h -> f16 (reuses ha16)
  k_norm_to_f16<<<(NH + TB - 1) / TB, blk, 0, stream>>>(ha, scale, shift, ha16, NH);
  // 12) logits = hbn@Wf2 + b_f2  -> d_out [N, 40]
  k_gemm_wmma<5, 3><<<gC, blk, 0, stream>>>(ha16, Wf2T, b_f2, nullptr, nullptr,
                                            out, nullptr, nullptr, nullptr,
                                            N, H_DIM, C_DIM, C_DIM);

  (void)n_in; (void)out_size; (void)ws_size; (void)off;
}
